// FuseBlock_61735859913347
// MI455X (gfx1250) — matrix operand order repaired and checked by reading, not compiled, not run on hardware
//
#include <hip/hip_runtime.h>
#include <hip/hip_bf16.h>
#include <stdint.h>

typedef __attribute__((ext_vector_type(16))) _Float16 v16h;
typedef __attribute__((ext_vector_type(8)))  _Float16 v8h;
typedef __attribute__((ext_vector_type(8)))  float    v8f;

// Problem constants (match reference): B=2, C=64, H=W=128, SCALE=2 -> H2=W2=256
// ---------------------------------------------------------------------------
// Layout conversion: NCHW f32 -> NHWC f16
// ---------------------------------------------------------------------------

// depth+guide concat: dst (B,256,256,128) f16, ch 0..63 = depth, 64..127 = guide
__global__ void pack_concat_kernel(const float* __restrict__ d,
                                   const float* __restrict__ g,
                                   _Float16* __restrict__ dst) {
  size_t i = (size_t)blockIdx.x * 256 + threadIdx.x;   // B*65536*128 threads
  int c = (int)(i & 127);
  size_t pix = i >> 7;           // b*65536 + hw
  size_t b  = pix >> 16;
  size_t hw = pix & 65535;
  const float* src = (c < 64) ? d : g;
  int cc = c & 63;
  dst[i] = (_Float16)src[((b * 64 + cc) << 16) + hw];
}

// generic NCHW(B,64,HW) f32 -> NHWC f16
__global__ void pack_act_kernel(const float* __restrict__ src,
                                _Float16* __restrict__ dst, int HW) {
  size_t i = (size_t)blockIdx.x * 256 + threadIdx.x;   // B*HW*64 threads
  int c = (int)(i & 63);
  size_t pix = i >> 6;
  size_t b  = pix / (size_t)HW;
  size_t hw = pix - b * (size_t)HW;
  dst[i] = (_Float16)src[(b * 64 + c) * (size_t)HW + hw];
}

// ---------------------------------------------------------------------------
// Weight pre-swizzle: OIHW f32 -> per-lane WMMA B-fragment image (f16 pairs).
// K ordering: k = tap*Cin + ic. Output dword index:
//   i = ((chunk*4 + ntile)*32 + lane)*8 + dword
// B layout (16-bit, K32 x N16): lane n<16 -> col n, K = 0..7,16..23 (2 per dword);
// lanes 16..31 -> col n-16, K = 8..15,24..31 of the chunk.
// ---------------------------------------------------------------------------
__global__ void pack_w_kernel(const float* __restrict__ w,
                              uint32_t* __restrict__ dst,
                              int Cin, int taps, int total) {
  int i = blockIdx.x * 256 + threadIdx.x;
  if (i >= total) return;
  int d    = i & 7;
  int lane = (i >> 3) & 31;
  int nt   = (i >> 8) & 3;
  int ch   = i >> 10;
  int oc   = (nt << 4) + (lane & 15);
  int kb   = (ch << 5) + ((lane >> 4) << 3) + (d << 1) + ((d >> 2) << 3);
  uint32_t outv = 0;
  for (int hh = 0; hh < 2; ++hh) {
    int k  = kb + hh;
    int t  = k / Cin;
    int ic = k - t * Cin;
    float v = w[((size_t)(oc * Cin + ic)) * taps + t];
    union { _Float16 h; uint16_t u; } cvt;
    cvt.h = (_Float16)v;
    outv |= ((uint32_t)cvt.u) << (16 * hh);
  }
  dst[i] = outv;
}

// ---------------------------------------------------------------------------
// Implicit-GEMM conv via v_wmma_f32_16x16x32_f16. Cout = 64 (4 N-tiles).
// Block = 256 threads = 8 waves; wave (mt,nt): 16 pixels x 16 out-channels.
// Block covers 32 consecutive pixels of one image row.
// CIN/TAPS/PITCH/RELU/HAS_SKIP are compile-time -> fully unrolled K loop with
// immediate tap offsets and a branch-free epilogue.
// ---------------------------------------------------------------------------
template <int CIN, int TAPS, int PITCH, bool RELU, bool HAS_SKIP>
__global__ __launch_bounds__(256)
void wmma_conv64_kernel(const _Float16* __restrict__ act, int chanOff,
                        const uint32_t* __restrict__ wpack,
                        const float* __restrict__ bias,
                        const _Float16* __restrict__ skip,   // NHWC f16 (64ch)
                        _Float16* __restrict__ out,          // NHWC f16 (64ch)
                        int H, int W) {
  const int tid  = threadIdx.x;
  const int wave = tid >> 5;
  const int lane = tid & 31;
  const int half = lane >> 4;     // K-half selector
  const int nrow = lane & 15;     // M row (for A) / N col (for B,C,D)
  const int mt = wave >> 2;       // 0..1 pixel sub-tile
  const int nt = wave & 3;        // 0..3 out-channel tile
  const int wtiles = W >> 5;
  int bidx = blockIdx.x;
  const int w0 = (bidx % wtiles) << 5;
  bidx /= wtiles;
  const int h = bidx % H;
  const int b = bidx / H;
  const int myw = w0 + (mt << 4) + nrow;

  // center-pixel activation pointer for this lane (K-half applied once)
  const _Float16* p0 = act + ((size_t)(b * H + h) * W + myw) * PITCH
                           + chanOff + (half << 3);
  // weight fragment pointer: chunk stride = 4 ntiles * 32 lanes * 8 dwords
  const uint32_t* wp = wpack + (((size_t)nt * 32 + lane) << 3);

  // hint the prefetcher at the next 32-pixel tile of this row
  __builtin_prefetch(p0 + (size_t)32 * PITCH, 0, 1);

  v8f acc = {};
  constexpr int NCH = (TAPS * CIN) >> 5;
#pragma unroll
  for (int ch = 0; ch < NCH; ++ch) {
    const int k0  = ch << 5;
    const int t   = k0 / CIN;
    const int icb = k0 - t * CIN;
    const int di  = (TAPS == 9) ? (t / 3 - 1) : 0;
    const int dj  = (TAPS == 9) ? (t % 3 - 1) : 0;
    v16h a = {};
    bool inb = (TAPS == 1) ||
               (((unsigned)(h + di) < (unsigned)H) &&
                ((unsigned)(myw + dj) < (unsigned)W));
    if (inb) {
      const _Float16* bse = p0 + ((di * W + dj) * PITCH + icb);
      union { v16h v; v8h p[2]; } ua;
      ua.p[0] = *(const v8h*)(bse);        // K {0..7}  (+8 for upper half)
      ua.p[1] = *(const v8h*)(bse + 16);   // K {16..23}(+8 for upper half)
      a = ua.v;
    }
    v16h bv = *(const v16h*)(wp + (size_t)ch * 1024);
    acc = __builtin_amdgcn_wmma_f32_16x16x32_f16(
        /*neg_a=*/false, a, /*neg_b=*/false, bv,
        /*c_mod=*/(short)0, acc, /*reuse_a=*/false, /*reuse_b=*/false);
  }

  const int oc = (nt << 4) + nrow;
  const float bs = bias[oc];
  const size_t rowpix = (size_t)(b * H + h) * W + (w0 + (mt << 4) + (half << 3));
#pragma unroll
  for (int r = 0; r < 8; ++r) {
    float v = acc[r] + bs;               // C/D layout: element r -> M = r + 8*half
    if constexpr (RELU) v = fmaxf(v, 0.f);
    if constexpr (HAS_SKIP) v += (float)skip[(rowpix + r) * 64 + oc];
    out[(rowpix + r) * 64 + oc] = (_Float16)v;
  }
}

// ---------------------------------------------------------------------------
// 1x1 conv 64->9 head + softmax over the 9 logits. One thread per pixel.
// ---------------------------------------------------------------------------
__global__ void head_softmax_kernel(const _Float16* __restrict__ hid,
                                    const float* __restrict__ w2,
                                    const float* __restrict__ b2,
                                    float* __restrict__ out9) {
  size_t p = (size_t)blockIdx.x * 256 + threadIdx.x;   // 131072 pixels
  const _Float16* hp = hid + p * 64;
  float acc[9];
#pragma unroll
  for (int k = 0; k < 9; ++k) acc[k] = b2[k];
  for (int ic = 0; ic < 64; ++ic) {
    float v = (float)hp[ic];
#pragma unroll
    for (int k = 0; k < 9; ++k) acc[k] = fmaf(v, w2[k * 64 + ic], acc[k]);
  }
  float mx = acc[0];
#pragma unroll
  for (int k = 1; k < 9; ++k) mx = fmaxf(mx, acc[k]);
  float s = 0.f;
#pragma unroll
  for (int k = 0; k < 9; ++k) { acc[k] = __expf(acc[k] - mx); s += acc[k]; }
  float inv = 1.f / s;
#pragma unroll
  for (int k = 0; k < 9; ++k) out9[p * 9 + k] = acc[k] * inv;
}

// 3x3 conv 64->1 (weight map). One thread per pixel @256x256.
__global__ void wmap_kernel(const _Float16* __restrict__ x,
                            const float* __restrict__ w3,
                            const float* __restrict__ b3,
                            float* __restrict__ wm) {
  size_t p = (size_t)blockIdx.x * 256 + threadIdx.x;
  int b = (int)(p >> 16);
  int h = (int)((p >> 8) & 255);
  int w = (int)(p & 255);
  float acc = b3[0];
#pragma unroll
  for (int t = 0; t < 9; ++t) {
    int hs = h + t / 3 - 1;
    int ws = w + t % 3 - 1;
    if ((unsigned)hs < 256u && (unsigned)ws < 256u) {
      const _Float16* bp = x + ((((size_t)b << 8) + hs) * 256 + ws) * 64;
      for (int ic = 0; ic < 64; ++ic)
        acc = fmaf((float)bp[ic], w3[ic * 9 + t], acc);
    }
  }
  wm[p] = acc;
}

// fuse = normalize(tanh(wm*dk + (1-wm)*gk) / aff)
__global__ void fuse_kernel(const float* __restrict__ dk,
                            const float* __restrict__ gk,
                            const float* __restrict__ wm,
                            const float* __restrict__ aff,
                            float* __restrict__ fo) {
  size_t p = (size_t)blockIdx.x * 256 + threadIdx.x;
  float w = wm[p];
  float inv_aff = 1.f / (aff[0] + 1e-8f);
  float f[9];
  float asum = 0.f;
#pragma unroll
  for (int k = 0; k < 9; ++k) {
    float v = w * dk[p * 9 + k] + (1.f - w) * gk[p * 9 + k];
    v = tanhf(v) * inv_aff;
    f[k] = v;
    asum += fabsf(v);
  }
  float inv = 1.f / fmaxf(asum + 1e-4f, 1.f);
#pragma unroll
  for (int k = 0; k < 9; ++k) fo[p * 9 + k] = f[k] * inv;
}

// bilinear x2 (half-pixel centers, edge clamp): (B,128,128,64)f16 -> (B,256,256,64)f16
__global__ void upsample_kernel(const _Float16* __restrict__ x,
                                _Float16* __restrict__ up) {
  size_t i = (size_t)blockIdx.x * 256 + threadIdx.x;   // B*65536*64
  int c  = (int)(i & 63);
  size_t pi = i >> 6;
  int wq = (int)(pi & 255);
  int hq = (int)((pi >> 8) & 255);
  int b  = (int)(pi >> 16);
  float fy = hq * 0.5f - 0.25f;
  float fx = wq * 0.5f - 0.25f;
  int y0 = (int)floorf(fy), x0 = (int)floorf(fx);
  float wy = fy - (float)y0, wx = fx - (float)x0;
  int y1 = y0 + 1 > 127 ? 127 : y0 + 1;
  int x1 = x0 + 1 > 127 ? 127 : x0 + 1;
  if (y0 < 0) y0 = 0;
  if (x0 < 0) x0 = 0;
  const _Float16* base = x + ((size_t)b << 20);        // 128*128*64 = 1<<20
  float v00 = (float)base[(((size_t)y0 << 7) + x0) * 64 + c];
  float v01 = (float)base[(((size_t)y0 << 7) + x1) * 64 + c];
  float v10 = (float)base[(((size_t)y1 << 7) + x0) * 64 + c];
  float v11 = (float)base[(((size_t)y1 << 7) + x1) * 64 + c];
  float v = (1.f - wy) * ((1.f - wx) * v00 + wx * v01)
          +        wy  * ((1.f - wx) * v10 + wx * v11);
  up[i] = (_Float16)v;
}

// out[b,c,h,w] = sum_t fuse[p,t] * up[b, h+2(ti-1), w+2(tj-1), c] (zero-pad)
__global__ void gather_kernel(const float* __restrict__ fuse9,
                              const _Float16* __restrict__ up,
                              float* __restrict__ out) {
  size_t i = (size_t)blockIdx.x * 256 + threadIdx.x;   // B*65536*64
  int c = (int)(i & 63);
  size_t pi = i >> 6;
  int w = (int)(pi & 255);
  int h = (int)((pi >> 8) & 255);
  int b = (int)(pi >> 16);
  const float* f = fuse9 + pi * 9;
  float acc = 0.f;
#pragma unroll
  for (int t = 0; t < 9; ++t) {
    int hs = h + ((t / 3) - 1) * 2;
    int ws = w + ((t % 3) - 1) * 2;
    if ((unsigned)hs < 256u && (unsigned)ws < 256u)
      acc = fmaf(f[t], (float)up[((((size_t)b << 8) + hs) * 256 + ws) * 64 + c], acc);
  }
  out[(((size_t)b * 64 + c) * 256 + h) * 256 + w] = acc;
}

// ---------------------------------------------------------------------------
extern "C" void kernel_launch(void* const* d_in, const int* in_sizes, int n_in,
                              void* d_out, int out_size, void* d_ws, size_t ws_size,
                              hipStream_t stream) {
  (void)in_sizes; (void)n_in; (void)out_size; (void)ws_size;
  const float* depth  = (const float*)d_in[0];
  const float* guide  = (const float*)d_in[1];
  const float* inputs = (const float*)d_in[2];
  const float* dk_w1  = (const float*)d_in[3];
  const float* dk_b1  = (const float*)d_in[4];
  const float* dk_w2  = (const float*)d_in[5];
  const float* dk_b2  = (const float*)d_in[6];
  const float* gk_w1  = (const float*)d_in[7];
  const float* gk_b1  = (const float*)d_in[8];
  const float* gk_w2  = (const float*)d_in[9];
  const float* gk_b2  = (const float*)d_in[10];
  const float* wn_w1  = (const float*)d_in[11];
  const float* wn_b1  = (const float*)d_in[12];
  const float* wn_w2  = (const float*)d_in[13];
  const float* wn_b2  = (const float*)d_in[14];
  const float* wn_w3  = (const float*)d_in[15];
  const float* wn_b3  = (const float*)d_in[16];
  const float* rb_w   = (const float*)d_in[17];
  const float* rb_b   = (const float*)d_in[18];
  const float* aff    = (const float*)d_in[19];
  float* out = (float*)d_out;

  char* ws = (char*)d_ws;
  size_t off = 0;
  auto carve = [&](size_t bytes) -> char* {
    char* p = ws + off;
    off += (bytes + 255) & ~(size_t)255;
    return p;
  };
  _Float16* actDG = (_Float16*)carve(33554432);  // (B,256,256,128) f16
  _Float16* actIn = (_Float16*)carve(4194304);   // (B,128,128,64)  f16 (later: x2)
  _Float16* bufH  = (_Float16*)carve(16777216);  // hidden / wm2    f16
  _Float16* bufW1 = (_Float16*)carve(16777216);  // wm1 / up        f16
  float*    dk9   = (float*)carve(4718592);      // (B*65536, 9)
  float*    gk9   = (float*)carve(4718592);
  float*    fuse9 = (float*)carve(4718592);
  float*    wmap  = (float*)carve(524288);       // (B*65536)
  _Float16* bufR  = (_Float16*)carve(4194304);
  _Float16* bufX1 = (_Float16*)carve(4194304);
  uint32_t* pk_wn1 = (uint32_t*)carve(147456);   // 36 chunks * 4 * 32 * 8 dwords
  uint32_t* pk_wn2 = (uint32_t*)carve(73728);    // 18 chunks
  uint32_t* pk_dk1 = (uint32_t*)carve(8192);     //  2 chunks
  uint32_t* pk_gk1 = (uint32_t*)carve(8192);
  uint32_t* pk_rb[4];
  for (int i = 0; i < 4; ++i) pk_rb[i] = (uint32_t*)carve(73728);

  // ---- layout conversions + weight swizzle ----
  pack_concat_kernel<<<65536, 256, 0, stream>>>(depth, guide, actDG);
  pack_act_kernel<<<8192, 256, 0, stream>>>(inputs, actIn, 16384);
  pack_w_kernel<<<144, 256, 0, stream>>>(wn_w1, pk_wn1, 128, 9, 36864);
  pack_w_kernel<<<72,  256, 0, stream>>>(wn_w2, pk_wn2, 64, 9, 18432);
  pack_w_kernel<<<8,   256, 0, stream>>>(dk_w1, pk_dk1, 64, 1, 2048);
  pack_w_kernel<<<8,   256, 0, stream>>>(gk_w1, pk_gk1, 64, 1, 2048);
  for (int i = 0; i < 4; ++i)
    pack_w_kernel<<<72, 256, 0, stream>>>(rb_w + (size_t)i * 64 * 64 * 9,
                                          pk_rb[i], 64, 9, 18432);

  const int gridHi = 2 * 256 * 8;   // B * H2 * (W2/32)
  const int gridLo = 2 * 128 * 4;   // B * H  * (W/32)

  // ---- dk / gk branches: 1x1 conv (WMMA) + head/softmax ----
  wmma_conv64_kernel<64, 1, 128, true, false><<<gridHi, 256, 0, stream>>>(
      actDG, 0, pk_dk1, dk_b1, nullptr, bufH, 256, 256);
  head_softmax_kernel<<<512, 256, 0, stream>>>(bufH, dk_w2, dk_b2, dk9);
  wmma_conv64_kernel<64, 1, 128, true, false><<<gridHi, 256, 0, stream>>>(
      actDG, 64, pk_gk1, gk_b1, nullptr, bufH, 256, 256);
  head_softmax_kernel<<<512, 256, 0, stream>>>(bufH, gk_w2, gk_b2, gk9);

  // ---- weight net: 3x3 convs (WMMA) + 64->1 head ----
  wmma_conv64_kernel<128, 9, 128, true, false><<<gridHi, 256, 0, stream>>>(
      actDG, 0, pk_wn1, wn_b1, nullptr, bufW1, 256, 256);
  wmma_conv64_kernel<64, 9, 64, true, false><<<gridHi, 256, 0, stream>>>(
      bufW1, 0, pk_wn2, wn_b2, nullptr, bufH, 256, 256);
  wmap_kernel<<<512, 256, 0, stream>>>(bufH, wn_w3, wn_b3, wmap);
  fuse_kernel<<<512, 256, 0, stream>>>(dk9, gk9, wmap, aff, fuse9);

  // ---- residual stack on low-res inputs (WMMA, fused skip-add) ----
  wmma_conv64_kernel<64, 9, 64, true, false><<<gridLo, 256, 0, stream>>>(
      actIn, 0, pk_rb[0], rb_b + 0, nullptr, bufR, 128, 128);
  wmma_conv64_kernel<64, 9, 64, false, true><<<gridLo, 256, 0, stream>>>(
      bufR, 0, pk_rb[1], rb_b + 64, actIn, bufX1, 128, 128);
  wmma_conv64_kernel<64, 9, 64, true, false><<<gridLo, 256, 0, stream>>>(
      bufX1, 0, pk_rb[2], rb_b + 128, nullptr, bufR, 128, 128);
  wmma_conv64_kernel<64, 9, 64, false, true><<<gridLo, 256, 0, stream>>>(
      bufR, 0, pk_rb[3], rb_b + 192, bufX1, actIn, 128, 128);

  // ---- upsample + fused dilated-unfold gather ----
  upsample_kernel<<<32768, 256, 0, stream>>>(actIn, bufW1);
  gather_kernel<<<32768, 256, 0, stream>>>(fuse9, bufW1, out);
}
